// BaselineBlockNetSingleGraph_4054449127563
// MI455X (gfx1250) — matrix operand order repaired and checked by reading, not compiled, not run on hardware
//
#include <hip/hip_runtime.h>
#include <hip/hip_bf16.h>

// Dims (fixed by reference): B=64, W=48, N=64, C=H=64, QK=32, HOR=12
#define BB 64
#define WW 48
#define NN 64
#define CC 64
#define HH 64
#define QK 32
#define HOR 12
#define THREEH 192
#define M_GRU (BB * NN)            // 4096 series
#define FDIM (WW * NN * CC)        // 196608 flattened features
#define ODIM (NN * HOR)            // 768 outputs per batch row

typedef __attribute__((ext_vector_type(16))) _Float16 v16h;
typedef __attribute__((ext_vector_type(8)))  float    v8f;

__device__ static inline v8f wmma16(v16h a, v16h b, v8f c) {
  // D = A(16x32 f16) * B(32x16 f16) + C(16x16 f32)
  return __builtin_amdgcn_wmma_f32_16x16x32_f16(
      /*neg_a=*/false, a, /*neg_b=*/false, b,
      /*c_mod=*/(short)0, c, /*reuse_a=*/false, /*reuse_b=*/false);
}

// Fast transcendental helpers (v_rcp_f32 / v_exp_f32 instead of IEEE div and
// ocml tanh libcall) — these sit on the serial GRU critical path.
__device__ static inline float fast_rcp(float x) { return __builtin_amdgcn_rcpf(x); }
__device__ static inline float fast_sigmoid(float x) {
  return fast_rcp(1.f + __expf(-x));
}
__device__ static inline float fast_tanh(float x) {
  // tanh(x) = 1 - 2/(exp(2x)+1)
  return 1.f - 2.f * fast_rcp(__expf(2.f * x) + 1.f);
}

// A-fragment (16x32, f16): lane l holds row m0+(l&15); halves 0..15 = K[kg..kg+15],
// kg = 16*(l>>4). Source row-major, stride 'ld'.
__device__ static inline v16h load_a(const _Float16* s, int ld, int m0, int k0, int lane) {
  const _Float16* p = s + (m0 + (lane & 15)) * ld + k0;
  int kg = 16 * (lane >> 4);
  v16h a;
#pragma unroll
  for (int i = 0; i < 8; ++i) { a[i] = p[kg + i]; a[8 + i] = p[8 + kg + i]; }
  return a;
}

// B-fragment from a weight matrix stored [n][k] (i.e. B[k][n] = w[n][k]):
__device__ static inline v16h load_b_wt(const _Float16* w, int ld, int k0, int n0, int lane) {
  const _Float16* p = w + (n0 + (lane & 15)) * ld + k0 + 16 * (lane >> 4);
  v16h b;
#pragma unroll
  for (int i = 0; i < 16; ++i) b[i] = p[i];
  return b;
}

// B-fragment from a matrix stored [k][n]:
__device__ static inline v16h load_b_kn(const _Float16* s, int ld, int k0, int n0, int lane) {
  const _Float16* p = s + (k0 + 16 * (lane >> 4)) * ld + (n0 + (lane & 15));
  v16h b;
#pragma unroll
  for (int i = 0; i < 16; ++i) b[i] = p[i * ld];
  return b;
}

// ---------------------------------------------------------------------------
// K1: GRU scan. 256 WGs x 384 threads (12 waves). Each WG owns 16 series rows;
// wave w owns gate columns [16w, 16w+16). W_hh resident in LDS, B-fragments
// resident in registers across all 48 steps. 2 WMMAs per wave per step.
// ---------------------------------------------------------------------------
__global__ __launch_bounds__(384)
void gru_kernel(const float* __restrict__ x, const float* __restrict__ w_ih,
                const float* __restrict__ w_hh, const float* __restrict__ b_ih,
                const float* __restrict__ b_hh, float* __restrict__ enc) {
  __shared__ _Float16 sWhh[THREEH * HH];   // [n][k] 24KB
  __shared__ float    sGh[16 * THREEH];    // 12KB
  __shared__ float    sH32[16 * HH];       // 4KB
  __shared__ _Float16 sH16[16 * HH];       // 2KB

  const int tid  = threadIdx.x;
  const int lane = tid & 31;
  const int wave = tid >> 5;               // 0..11 = gate n-tile
  const int row0 = blockIdx.x * 16;

  for (int i = tid; i < THREEH * HH; i += 384) sWhh[i] = (_Float16)w_hh[i];
  for (int i = tid; i < 16 * HH; i += 384) { sH32[i] = 0.f; sH16[i] = (_Float16)0.f; }
  __syncthreads();

  // Register-resident B fragments (K = 0..31 and 32..63) for this wave's tile.
  const v16h bf0 = load_b_wt(sWhh, HH, 0,  wave * 16, lane);
  const v16h bf1 = load_b_wt(sWhh, HH, 32, wave * 16, lane);

  for (int t = 0; t < WW; ++t) {
    v8f acc = {};
    v16h a0 = load_a(sH16, HH, 0, 0,  lane);
    v16h a1 = load_a(sH16, HH, 0, 32, lane);
    acc = wmma16(a0, bf0, acc);
    acc = wmma16(a1, bf1, acc);
    const int n = wave * 16 + (lane & 15);
#pragma unroll
    for (int r = 0; r < 8; ++r) {
      int m = r + 8 * (lane >> 4);
      sGh[m * THREEH + n] = acc[r];
    }
    __syncthreads();
    // Gates: 16 rows x 64 hidden
    for (int e = tid; e < 16 * HH; e += 384) {
      int m = e >> 6, j = e & 63;
      int row = row0 + m;                       // row = b*N + n
      int bidx = row >> 6, nidx = row & 63;
      float xv = x[(bidx * WW + t) * NN + nidx];
      float ir  = xv * w_ih[j]         + b_ih[j];
      float iz  = xv * w_ih[HH + j]    + b_ih[HH + j];
      float in_ = xv * w_ih[2*HH + j]  + b_ih[2*HH + j];
      float hr = sGh[m * THREEH + j]          + b_hh[j];
      float hz = sGh[m * THREEH + HH + j]     + b_hh[HH + j];
      float hn = sGh[m * THREEH + 2*HH + j]   + b_hh[2*HH + j];
      float rg = fast_sigmoid(ir + hr);
      float zg = fast_sigmoid(iz + hz);
      float ng = fast_tanh(in_ + rg * hn);
      float hnew = (1.f - zg) * ng + zg * sH32[e];
      sH32[e] = hnew;
      sH16[e] = (_Float16)hnew;
    }
    __syncthreads();
  }
  for (int e = tid; e < 16 * HH; e += 384)
    enc[(size_t)(row0 + (e >> 6)) * HH + (e & 63)] = sH32[e];
}

// ---------------------------------------------------------------------------
// K2: attention + gcn_norm. One WG per batch. Writes Anorm^T (f16) [b][j][i].
// Tiny (<<1% FLOPs) -> plain VALU.
// ---------------------------------------------------------------------------
__global__ __launch_bounds__(256)
void attn_kernel(const float* __restrict__ enc, const float* __restrict__ wq_w,
                 const float* __restrict__ wq_b, const float* __restrict__ wk_w,
                 const float* __restrict__ wk_b, _Float16* __restrict__ AnT) {
  __shared__ float sQ[NN * QK];
  __shared__ float sK[NN * QK];
  __shared__ float sS[NN * NN];
  __shared__ float sDi[NN];
  const int b = blockIdx.x;
  const int tid = threadIdx.x;
  const float* e = enc + (size_t)b * NN * HH;

  for (int idx = tid; idx < NN * QK; idx += 256) {
    int i = idx >> 5, q = idx & 31;
    float aq = wq_b[q], ak = wk_b[q];
    for (int h = 0; h < HH; ++h) {
      float ev = e[i * HH + h];
      aq += ev * wq_w[q * HH + h];
      ak += ev * wk_w[q * HH + h];
    }
    sQ[idx] = aq; sK[idx] = ak;
  }
  __syncthreads();
  for (int idx = tid; idx < NN * NN; idx += 256) {
    int i = idx >> 6, j = idx & 63;
    float s = 0.f;
    for (int q = 0; q < QK; ++q) s += sQ[i * QK + q] * sK[j * QK + q];
    sS[idx] = s * 0.17677669529663687f;   // 1/sqrt(32)
  }
  __syncthreads();
  if (tid < NN) {                          // row softmax over j
    int i = tid;
    float mx = -1e30f;
    for (int j = 0; j < NN; ++j) mx = fmaxf(mx, sS[i * NN + j]);
    float sum = 0.f;
    for (int j = 0; j < NN; ++j) { float v = __expf(sS[i * NN + j] - mx); sS[i * NN + j] = v; sum += v; }
    float inv = fast_rcp(sum);
    for (int j = 0; j < NN; ++j) sS[i * NN + j] *= inv;
  }
  __syncthreads();
  if (tid < NN) {                          // deg = column sums; dinv = deg^-1/2
    int j = tid; float s = 0.f;
    for (int i = 0; i < NN; ++i) s += sS[i * NN + j];
    sDi[j] = rsqrtf(s);
  }
  __syncthreads();
  for (int idx = tid; idx < NN * NN; idx += 256) {
    int j = idx >> 6, i = idx & 63;        // store transposed: [j][i]
    AnT[(size_t)b * NN * NN + idx] = (_Float16)(sDi[i] * sS[i * NN + j] * sDi[j]);
  }
}

// ---------------------------------------------------------------------------
// K3: embedding (pointwise 1->C conv), f32 -> f16 intermediate.
// ---------------------------------------------------------------------------
__global__ void embed_kernel(const float* __restrict__ x, const float* __restrict__ emb_w,
                             const float* __restrict__ emb_b, _Float16* __restrict__ h) {
  int idx = blockIdx.x * blockDim.x + threadIdx.x;
  if (idx >= BB * WW * NN * CC) return;
  int c = idx & 63;
  h[idx] = (_Float16)(x[idx >> 6] * emb_w[c] + emb_b[c]);
}

// ---------------------------------------------------------------------------
// K4: fused GCN linear + neighbor aggregation per (b,t).
// g = h(b,t) @ gw^T ; out = Anorm^T(b) @ g + gb. Two 64^3 WMMA GEMMs via LDS.
// 3072 WGs x 256 threads (8 waves, 2 tiles each of the 4x4 tile grid).
// ---------------------------------------------------------------------------
__global__ __launch_bounds__(256)
void gcn_agg_kernel(const _Float16* __restrict__ h, const _Float16* __restrict__ AnT,
                    const float* __restrict__ gw, const float* __restrict__ gb,
                    _Float16* __restrict__ gout) {
  __shared__ _Float16 sIn[CC * CC];
  __shared__ _Float16 sW[CC * CC];
  __shared__ _Float16 sG[CC * CC];
  __shared__ _Float16 sA[CC * CC];
  const int bt = blockIdx.x;
  const int b = bt / WW;
  const int tid = threadIdx.x, lane = tid & 31, wave = tid >> 5;
  const _Float16* hin = h + (size_t)bt * (NN * CC);
  for (int i = tid; i < CC * CC; i += 256) {
    sIn[i] = hin[i];
    sW[i]  = (_Float16)gw[i];
    sA[i]  = AnT[(size_t)b * NN * NN + i];
  }
  __syncthreads();
  for (int tile = wave; tile < 16; tile += 8) {          // GEMM1: g[n][c]
    int mt = tile >> 2, nt = tile & 3;
    v8f acc = {};
#pragma unroll
    for (int kb = 0; kb < 2; ++kb) {
      v16h a  = load_a(sIn, CC, mt * 16, kb * 32, lane);
      v16h bb = load_b_wt(sW, CC, kb * 32, nt * 16, lane);
      acc = wmma16(a, bb, acc);
    }
    int n = nt * 16 + (lane & 15);
#pragma unroll
    for (int r = 0; r < 8; ++r)
      sG[(mt * 16 + r + 8 * (lane >> 4)) * CC + n] = (_Float16)acc[r];
  }
  __syncthreads();
  _Float16* outp = gout + (size_t)bt * (NN * CC);
  for (int tile = wave; tile < 16; tile += 8) {          // GEMM2: out[j][c]
    int mt = tile >> 2, nt = tile & 3;
    v8f acc = {};
#pragma unroll
    for (int kb = 0; kb < 2; ++kb) {
      v16h a  = load_a(sA, NN, mt * 16, kb * 32, lane);
      v16h bb = load_b_kn(sG, CC, kb * 32, nt * 16, lane);
      acc = wmma16(a, bb, acc);
    }
    int n = nt * 16 + (lane & 15);
#pragma unroll
    for (int r = 0; r < 8; ++r)
      outp[(mt * 16 + r + 8 * (lane >> 4)) * CC + n] = (_Float16)(acc[r] + gb[n]);
  }
}

// ---------------------------------------------------------------------------
// K5: temporal conv (k = 3/5/7) + leaky_relu, as k accumulated CxC WMMA GEMMs
// over a zero-padded time slab in LDS. 4096 WGs (one per (b,n)) x 384 threads;
// wave tile = (t-tile 0..2, oc-tile 0..3).
// ---------------------------------------------------------------------------
__global__ __launch_bounds__(384)
void conv_kernel(const _Float16* __restrict__ gin, const float* __restrict__ cw,
                 const float* __restrict__ cb, _Float16* __restrict__ hout, int ksz) {
  const int PAD = 3;                         // max k//2
  __shared__ _Float16 sSlab[(WW + 2 * 3) * CC];  // 54 x 64 halves
  const int bn = blockIdx.x;
  const int b = bn >> 6, n = bn & 63;
  const int tid = threadIdx.x, lane = tid & 31, wave = tid >> 5;
  const int pad = ksz >> 1;

  for (int i = tid; i < (WW + 6) * CC; i += 384) sSlab[i] = (_Float16)0.f;
  __syncthreads();
  for (int i = tid; i < WW * CC; i += 384) {
    int t = i >> 6, c = i & 63;
    sSlab[(t + PAD) * CC + c] = gin[((size_t)(b * WW + t) * NN + n) * CC + c];
  }
  __syncthreads();

  const int mt = wave >> 2, nt = wave & 3;   // 3 x 4 tiles, one per wave
  const int oc = nt * 16 + (lane & 15);
  const int icb = 16 * (lane >> 4);
  const int kg = 16 * (lane >> 4);
  v8f acc = {};
  for (int dt = 0; dt < ksz; ++dt) {
    v16h b0, b1;
#pragma unroll
    for (int i = 0; i < 16; ++i) {
      b0[i] = (_Float16)cw[(oc * CC + (icb + i)) * ksz + dt];
      b1[i] = (_Float16)cw[(oc * CC + (32 + icb + i)) * ksz + dt];
    }
    const _Float16* ap = sSlab + (mt * 16 + (lane & 15) + dt - pad + PAD) * CC;
    v16h a0, a1;
#pragma unroll
    for (int i = 0; i < 8; ++i) {
      a0[i] = ap[kg + i];      a0[8 + i] = ap[8 + kg + i];
      a1[i] = ap[32 + kg + i]; a1[8 + i] = ap[40 + kg + i];
    }
    acc = wmma16(a0, b0, acc);
    acc = wmma16(a1, b1, acc);
  }
  const float bias = cb[oc];
#pragma unroll
  for (int r = 0; r < 8; ++r) {
    int t = mt * 16 + r + 8 * (lane >> 4);
    float v = acc[r] + bias;
    v = v > 0.f ? v : 0.01f * v;             // leaky_relu
    hout[((size_t)(b * WW + t) * NN + n) * CC + oc] = (_Float16)v;
  }
}

// ---------------------------------------------------------------------------
// K6: output bias init, then K-split output GEMM (K split over the 48 time
// chunks; fp32 global atomics). lout_w (604 MB fp32) streamed once, converted
// to f16 in registers. grid (48 n-tiles, 48 K-chunks) x 128 threads (4 waves
// = 4 m-tiles covering B=64 rows).
// ---------------------------------------------------------------------------
__global__ void out_init_kernel(const float* __restrict__ lout_b, float* __restrict__ out) {
  int idx = blockIdx.x * blockDim.x + threadIdx.x;
  if (idx < BB * ODIM) out[idx] = lout_b[idx % ODIM];
}

__global__ __launch_bounds__(128)
void lout_kernel(const _Float16* __restrict__ h, const float* __restrict__ lw,
                 float* __restrict__ out) {
  const int ntile = blockIdx.x;
  const int tchunk = blockIdx.y;
  const int lane = threadIdx.x & 31, wave = threadIdx.x >> 5;
  const int m0 = wave * 16;
  const size_t kbase = (size_t)tchunk * (NN * CC);       // 4096 per chunk
  const int nn = ntile * 16 + (lane & 15);
  const float* lwrow = lw + (size_t)nn * FDIM + kbase + 16 * (lane >> 4);
  const _Float16* hrow = h + (size_t)(m0 + (lane & 15)) * FDIM + kbase;
  const int kg = 16 * (lane >> 4);
  v8f acc = {};
  for (int kb = 0; kb < 128; ++kb) {
    const int k0 = kb * 32;
    __builtin_prefetch(lwrow + k0 + 256, 0, 1);          // global_prefetch_b8
    v16h a, bb;
#pragma unroll
    for (int i = 0; i < 8; ++i) { a[i] = hrow[k0 + kg + i]; a[8 + i] = hrow[k0 + 8 + kg + i]; }
#pragma unroll
    for (int i = 0; i < 16; ++i) bb[i] = (_Float16)lwrow[k0 + i];
    acc = wmma16(a, bb, acc);
  }
#pragma unroll
  for (int r = 0; r < 8; ++r) {
    int m = m0 + r + 8 * (lane >> 4);
    atomicAdd(&out[(size_t)m * ODIM + (ntile * 16 + (lane & 15))], acc[r]);
  }
}

// ---------------------------------------------------------------------------
extern "C" void kernel_launch(void* const* d_in, const int* in_sizes, int n_in,
                              void* d_out, int out_size, void* d_ws, size_t ws_size,
                              hipStream_t stream) {
  (void)in_sizes; (void)n_in; (void)out_size; (void)ws_size;
  const float* x      = (const float*)d_in[0];
  const float* w_ih   = (const float*)d_in[1];
  const float* w_hh   = (const float*)d_in[2];
  const float* b_ih   = (const float*)d_in[3];
  const float* b_hh   = (const float*)d_in[4];
  const float* wq_w   = (const float*)d_in[5];
  const float* wq_b   = (const float*)d_in[6];
  const float* wk_w   = (const float*)d_in[7];
  const float* wk_b   = (const float*)d_in[8];
  const float* emb_w  = (const float*)d_in[9];
  const float* emb_b  = (const float*)d_in[10];
  const float* lout_w = (const float*)d_in[23];
  const float* lout_b = (const float*)d_in[24];

  // Workspace layout (~49.5 MB total)
  char* ws = (char*)d_ws;
  float*    enc  = (float*)ws;                              // 4096*64*4   = 1 MB
  _Float16* AnT  = (_Float16*)(ws + (1u << 20));            // 64*4096*2   = 0.5 MB
  _Float16* hbuf = (_Float16*)(ws + 1572864);               // 12.58M * 2  = 25.2 MB
  _Float16* gbuf = (_Float16*)(ws + 26738688);              // 12.58M * 2  = 25.2 MB

  gru_kernel<<<M_GRU / 16, 384, 0, stream>>>(x, w_ih, w_hh, b_ih, b_hh, enc);
  attn_kernel<<<BB, 256, 0, stream>>>(enc, wq_w, wq_b, wk_w, wk_b, AnT);
  embed_kernel<<<(BB * WW * NN * CC + 255) / 256, 256, 0, stream>>>(x, emb_w, emb_b, hbuf);

  for (int blk = 0; blk < 3; ++blk) {
    const float* gw = (const float*)d_in[11 + blk * 4];
    const float* gb = (const float*)d_in[12 + blk * 4];
    const float* cw = (const float*)d_in[13 + blk * 4];
    const float* cb = (const float*)d_in[14 + blk * 4];
    const int ksz = 3 + 2 * blk;
    gcn_agg_kernel<<<BB * WW, 256, 0, stream>>>(hbuf, AnT, gw, gb, gbuf);
    conv_kernel<<<BB * NN, 384, 0, stream>>>(gbuf, cw, cb, hbuf, ksz);
  }

  out_init_kernel<<<(BB * ODIM + 255) / 256, 256, 0, stream>>>(lout_b, (float*)d_out);
  dim3 g2(ODIM / 16, WW);
  lout_kernel<<<g2, 128, 0, stream>>>(hbuf, lout_w, (float*)d_out);
}
